// SelectE_34608846471375
// MI455X (gfx1250) — matrix operand order repaired and checked by reading, not compiled, not run on hardware
//
#include <hip/hip_runtime.h>
#include <hip/hip_bf16.h>

#define B_DIM  1024
#define D_DIM  200
#define V_SIZE 100000
#define KPAD   224
#define EPSV   1e-5f

typedef __attribute__((ext_vector_type(16))) _Float16 v16h;
typedef __attribute__((ext_vector_type(8)))  float    v8f;

__constant__ float HAAR_LO[2] = {0.7071067811865476f, 0.7071067811865476f};
__constant__ float HAAR_HI[2] = {-0.7071067811865476f, 0.7071067811865476f};
__constant__ float DB2_LO[4]  = {-0.12940952255092145f, 0.22414386804185735f, 0.836516303737469f, 0.48296291314469025f};
__constant__ float DB2_HI[4]  = {-0.48296291314469025f, 0.836516303737469f, -0.22414386804185735f, -0.12940952255092145f};

__device__ __forceinline__ float sigmf(float x) { return 1.0f / (1.0f + __expf(-x)); }

// ---------------------------------------------------------------- prologue ---

// x_raw[b, j] = comb[b, perm[j]] ; comb = [emb[e1[b]], emb[rel[b]]]
__global__ void gather_kernel(const int* __restrict__ e1, const int* __restrict__ rel,
                              const int* __restrict__ perm, const float* __restrict__ emb,
                              float* __restrict__ xraw) {
  int b = blockIdx.x;
  int ie = e1[b], ir = rel[b];
  for (int j = threadIdx.x; j < 400; j += blockDim.x) {
    int p = perm[j];
    float v = (p < D_DIM) ? emb[(size_t)ie * D_DIM + p]
                          : emb[(size_t)ir * D_DIM + (p - D_DIM)];
    xraw[(size_t)b * 400 + j] = v;
  }
}

// global (1-channel) BN stats over all B*400 elements
__global__ void stats1_kernel(const float* __restrict__ x, float* __restrict__ stats, int n) {
  __shared__ float ss[256], sq[256];
  float s = 0.f, q = 0.f;
  for (int i = threadIdx.x; i < n; i += 256) { float v = x[i]; s += v; q += v * v; }
  ss[threadIdx.x] = s; sq[threadIdx.x] = q; __syncthreads();
  for (int o = 128; o > 0; o >>= 1) {
    if (threadIdx.x < o) { ss[threadIdx.x] += ss[threadIdx.x + o]; sq[threadIdx.x] += sq[threadIdx.x + o]; }
    __syncthreads();
  }
  if (threadIdx.x == 0) {
    float m = ss[0] / (float)n;
    float v = sq[0] / (float)n - m * m;
    stats[0] = m; stats[1] = rsqrtf(v + EPSV);
  }
}

// dwt2d per branch: h[br][b][c][u][v], stride-2 conv with pad p=(L-2)/2
__global__ void dwt_kernel(const float* __restrict__ xraw, const float* __restrict__ stats,
                           float* __restrict__ hbuf) {
  int b = blockIdx.x, br = blockIdx.y;
  float mean = stats[0], inv = stats[1];
  const float *lo, *hi; int L, p;
  if (br == 2) { lo = DB2_LO; hi = DB2_HI; L = 4; p = 1; }
  else         { lo = HAAR_LO; hi = HAAR_HI; L = 2; p = 0; }
  const float* xb = xraw + (size_t)b * 400;
  for (int idx = threadIdx.x; idx < 400; idx += blockDim.x) {
    int c = idx / 100, s = idx % 100, u = s / 10, v = s % 10;
    const float* fr = (c < 2) ? lo : hi;
    const float* fc = (c & 1) ? hi : lo;
    float acc = 0.f;
    for (int i = 0; i < L; ++i) {
      int r = 2 * u + i - p; if (r < 0 || r >= 20) continue;
      for (int j = 0; j < L; ++j) {
        int cc = 2 * v + j - p; if (cc < 0 || cc >= 20) continue;
        acc += fr[i] * fc[j] * ((xb[r * 20 + cc] - mean) * inv);
      }
    }
    hbuf[((size_t)(br * B_DIM + b) * 4 + c) * 100 + s] = acc;
  }
}

// per-(branch,channel) BN stats over B*100
__global__ void hstats_kernel(const float* __restrict__ hbuf, float* __restrict__ hstats) {
  int blk = blockIdx.x, br = blk >> 2, c = blk & 3;
  __shared__ float ss[256], sq[256];
  float s = 0.f, q = 0.f;
  for (int i = threadIdx.x; i < B_DIM * 100; i += 256) {
    int b = i / 100, sp = i % 100;
    float v = hbuf[((size_t)(br * B_DIM + b) * 4 + c) * 100 + sp];
    s += v; q += v * v;
  }
  ss[threadIdx.x] = s; sq[threadIdx.x] = q; __syncthreads();
  for (int o = 128; o > 0; o >>= 1) {
    if (threadIdx.x < o) { ss[threadIdx.x] += ss[threadIdx.x + o]; sq[threadIdx.x] += sq[threadIdx.x + o]; }
    __syncthreads();
  }
  if (threadIdx.x == 0) {
    float n = (float)(B_DIM * 100);
    float m = ss[0] / n;
    float v = sq[0] / n - m * m;
    hstats[blk * 2] = m; hstats[blk * 2 + 1] = rsqrtf(v + EPSV);
  }
}

// BN-apply + squeeze-excite + idwt2d (adjoint of the stride-2 conv)
__global__ void se_idwt_kernel(const float* __restrict__ hbuf, const float* __restrict__ hstats,
                               const float* __restrict__ se_w1, const float* __restrict__ se_b1,
                               const float* __restrict__ se_w2, const float* __restrict__ se_b2,
                               float* __restrict__ xo) {
  int b = blockIdx.x, br = blockIdx.y, tid = threadIdx.x; // 128 threads
  __shared__ float hsh[400];
  __shared__ float red[4][128];
  __shared__ float aatt[4];
  const float* hb = hbuf + ((size_t)(br * B_DIM + b) * 4) * 100;
  float psum[4] = {0.f, 0.f, 0.f, 0.f};
  for (int idx = tid; idx < 400; idx += 128) {
    int c = idx / 100, s = idx % 100;
    float m = hstats[(br * 4 + c) * 2], inv = hstats[(br * 4 + c) * 2 + 1];
    float hn = (hb[c * 100 + s] - m) * inv;
    hsh[idx] = hn;
    psum[c] += hn;
  }
  for (int c = 0; c < 4; ++c) red[c][tid] = psum[c];
  __syncthreads();
  for (int o = 64; o > 0; o >>= 1) {
    if (tid < o) for (int c = 0; c < 4; ++c) red[c][tid] += red[c][tid + o];
    __syncthreads();
  }
  if (tid == 0) {
    float h1 = se_b1[br];
    for (int c = 0; c < 4; ++c) h1 += (red[c][0] * 0.01f) * se_w1[br * 4 + c];
    h1 = fmaxf(h1, 0.f);
    for (int c = 0; c < 4; ++c) aatt[c] = sigmf(h1 * se_w2[br * 4 + c] + se_b2[br * 4 + c]);
  }
  __syncthreads();
  const float *lo, *hi; int L, p;
  if (br == 2) { lo = DB2_LO; hi = DB2_HI; L = 4; p = 1; }
  else         { lo = HAAR_LO; hi = HAAR_HI; L = 2; p = 0; }
  for (int idx = tid; idx < 400; idx += 128) {
    int m = idx / 20, n = idx % 20;
    int lom = m + p - L + 1; int u0 = (lom <= 0) ? 0 : ((lom + 1) >> 1);
    int u1 = (m + p) >> 1; if (u1 > 9) u1 = 9;
    int lon = n + p - L + 1; int v0 = (lon <= 0) ? 0 : ((lon + 1) >> 1);
    int v1 = (n + p) >> 1; if (v1 > 9) v1 = 9;
    float acc = 0.f;
    for (int c = 0; c < 4; ++c) {
      const float* fr = (c < 2) ? lo : hi;
      const float* fc = (c & 1) ? hi : lo;
      float ac = aatt[c];
      for (int u = u0; u <= u1; ++u) {
        float kr = fr[m - 2 * u + p];
        for (int v = v0; v <= v1; ++v)
          acc += ac * kr * fc[n - 2 * v + p] * hsh[c * 100 + u * 10 + v];
      }
    }
    xo[((size_t)br * B_DIM + b) * 400 + idx] = acc;
  }
}

// PAG fusion (twice) + FC (400 -> 200)
__global__ void pag_fc_kernel(const float* __restrict__ xo, const float* __restrict__ pag_w,
                              const float* __restrict__ fc_w, const float* __restrict__ fc_b,
                              float* __restrict__ fc_out) {
  int b = blockIdx.x, tid = threadIdx.x; // 256 threads
  __shared__ float xfs[400];
  float w0 = pag_w[0], w1 = pag_w[1], w2 = pag_w[2], w3 = pag_w[3];
  for (int j = tid; j < 400; j += 256) {
    float x1 = xo[(size_t)b * 400 + j];
    float x3 = xo[((size_t)B_DIM + b) * 400 + j];
    float x5 = xo[((size_t)2 * B_DIM + b) * 400 + j];
    float s1 = sigmf((w0 * x1) * (w1 * x3));
    float m  = s1 * x3 + (1.f - s1) * x1;
    float s2 = sigmf((w2 * m) * (w3 * x5));
    xfs[j]   = s2 * x5 + (1.f - s2) * m;
  }
  __syncthreads();
  for (int d = tid; d < 200; d += 256) {
    float acc = fc_b[d];
    for (int j = 0; j < 400; ++j) acc += xfs[j] * fc_w[j * 200 + d];
    fc_out[(size_t)b * 200 + d] = acc;
  }
}

// per-feature BN stats over the batch (200 features x 1024 samples)
__global__ void fstats_kernel(const float* __restrict__ fc_out, float* __restrict__ fstats) {
  int d = blockIdx.x;
  __shared__ float ss[256], sq[256];
  float s = 0.f, q = 0.f;
  for (int b = threadIdx.x; b < B_DIM; b += 256) {
    float v = fc_out[(size_t)b * 200 + d]; s += v; q += v * v;
  }
  ss[threadIdx.x] = s; sq[threadIdx.x] = q; __syncthreads();
  for (int o = 128; o > 0; o >>= 1) {
    if (threadIdx.x < o) { ss[threadIdx.x] += ss[threadIdx.x + o]; sq[threadIdx.x] += sq[threadIdx.x + o]; }
    __syncthreads();
  }
  if (threadIdx.x == 0) {
    float m = ss[0] / (float)B_DIM;
    float v = sq[0] / (float)B_DIM - m * m;
    fstats[d * 2] = m; fstats[d * 2 + 1] = rsqrtf(v + EPSV);
  }
}

// relu(bn(fc_out)) -> f16 A matrix, K-padded 200 -> 224 with zeros
__global__ void prep_a_kernel(const float* __restrict__ fc_out, const float* __restrict__ fstats,
                              _Float16* __restrict__ A) {
  int i = blockIdx.x * 256 + threadIdx.x;
  if (i >= B_DIM * KPAD) return;
  int b = i / KPAD, d = i % KPAD;
  float v = 0.f;
  if (d < D_DIM) {
    float x = fc_out[(size_t)b * D_DIM + d];
    v = fmaxf((x - fstats[d * 2]) * fstats[d * 2 + 1], 0.f);
  }
  A[i] = (_Float16)v;
}

// ------------------------------------------------------------ WMMA GEMM -----
// out[b, v] = sum_d A[b,d] * emb[v,d] + bias[v]
// block = 256 threads (8 waves); tile M=64 (4 wmma m-tiles), N=128 (16 per wave)
__global__ __launch_bounds__(256) void gemm_kernel(const _Float16* __restrict__ A,
                                                   const float* __restrict__ emb,
                                                   const float* __restrict__ bias,
                                                   float* __restrict__ out) {
  __shared__ _Float16 As[64 * KPAD]; // 28 KB
  const int tid  = threadIdx.x;
  const int wave = tid >> 5;
  const int lane = tid & 31;
  const int half = lane >> 4;   // 0 or 1
  const int l16  = lane & 15;
  const int mBase = blockIdx.y * 64;
  const int nCol  = blockIdx.x * 128 + wave * 16 + l16;
  const bool colOK = (nCol < V_SIZE);       // wave-uniform: V % 16 == 0

  // cooperative A-tile load (rows are contiguous in global)
  {
    const uint4* src = (const uint4*)(A + (size_t)mBase * KPAD);
    uint4* dst = (uint4*)As;
    #pragma unroll
    for (int i = tid; i < (64 * KPAD) / 8; i += 256) dst[i] = src[i];
  }
  __syncthreads();

  v8f c0 = {0,0,0,0,0,0,0,0}, c1 = c0, c2 = c0, c3 = c0;

  const float* brow = emb + (size_t)(colOK ? nCol : 0) * D_DIM;

  union AF { v16h v; uint4 q[2]; };
  union BF { v16h v; _Float16 h[16]; };

  // main K loop: k0 = 0..160, all B elements in-range (k0+31 < 200)
  #pragma unroll
  for (int k0 = 0; k0 < 192; k0 += 32) {
    // B fragment: lane half h holds K {k0+8h .. k0+8h+7} and {k0+8h+16 .. +23}
    BF bf;
    {
      int kbs = k0 + half * 8;
      float4 f0 = *(const float4*)(brow + kbs);
      float4 f1 = *(const float4*)(brow + kbs + 4);
      float4 f2 = *(const float4*)(brow + kbs + 16);
      float4 f3 = *(const float4*)(brow + kbs + 20);
      bf.h[0]  = (_Float16)f0.x; bf.h[1]  = (_Float16)f0.y; bf.h[2]  = (_Float16)f0.z; bf.h[3]  = (_Float16)f0.w;
      bf.h[4]  = (_Float16)f1.x; bf.h[5]  = (_Float16)f1.y; bf.h[6]  = (_Float16)f1.z; bf.h[7]  = (_Float16)f1.w;
      bf.h[8]  = (_Float16)f2.x; bf.h[9]  = (_Float16)f2.y; bf.h[10] = (_Float16)f2.z; bf.h[11] = (_Float16)f2.w;
      bf.h[12] = (_Float16)f3.x; bf.h[13] = (_Float16)f3.y; bf.h[14] = (_Float16)f3.z; bf.h[15] = (_Float16)f3.w;
    }
    // A fragments: lane half h holds K {k0+8h .. k0+8h+7} and {k0+8h+16 .. +23}
    // Preload ALL four m-tile fragments (8 x ds_load_b128) before the WMMA burst
    // so the DS round-trips overlap and the 4 WMMAs issue back-to-back.
    int kb = k0 + half * 8;
    AF af0, af1, af2, af3;
    {
      int o0 = (0 * 16 + l16) * KPAD + kb;
      int o1 = (1 * 16 + l16) * KPAD + kb;
      int o2 = (2 * 16 + l16) * KPAD + kb;
      int o3 = (3 * 16 + l16) * KPAD + kb;
      af0.q[0] = *(const uint4*)(As + o0); af0.q[1] = *(const uint4*)(As + o0 + 16);
      af1.q[0] = *(const uint4*)(As + o1); af1.q[1] = *(const uint4*)(As + o1 + 16);
      af2.q[0] = *(const uint4*)(As + o2); af2.q[1] = *(const uint4*)(As + o2 + 16);
      af3.q[0] = *(const uint4*)(As + o3); af3.q[1] = *(const uint4*)(As + o3 + 16);
    }
    c0 = __builtin_amdgcn_wmma_f32_16x16x32_f16(false, af0.v, false, bf.v, (short)0, c0, false, false);
    c1 = __builtin_amdgcn_wmma_f32_16x16x32_f16(false, af1.v, false, bf.v, (short)0, c1, false, false);
    c2 = __builtin_amdgcn_wmma_f32_16x16x32_f16(false, af2.v, false, bf.v, (short)0, c2, false, false);
    c3 = __builtin_amdgcn_wmma_f32_16x16x32_f16(false, af3.v, false, bf.v, (short)0, c3, false, false);
  }

  // K tail: k0 = 192, valid K only in [192, 200)
  {
    const int k0 = 192;
    BF bf;
    int kbs = k0 + half * 8;
    #pragma unroll
    for (int i = 0; i < 16; ++i) {
      int k = kbs + i + (i >= 8 ? 8 : 0);
      bf.h[i] = (k < D_DIM) ? (_Float16)brow[k] : (_Float16)0.f;
    }
    int kb = k0 + half * 8;
    AF af0, af1, af2, af3;
    {
      int o0 = (0 * 16 + l16) * KPAD + kb; // A is zero-padded to 224, loads safe
      int o1 = (1 * 16 + l16) * KPAD + kb;
      int o2 = (2 * 16 + l16) * KPAD + kb;
      int o3 = (3 * 16 + l16) * KPAD + kb;
      af0.q[0] = *(const uint4*)(As + o0); af0.q[1] = *(const uint4*)(As + o0 + 16);
      af1.q[0] = *(const uint4*)(As + o1); af1.q[1] = *(const uint4*)(As + o1 + 16);
      af2.q[0] = *(const uint4*)(As + o2); af2.q[1] = *(const uint4*)(As + o2 + 16);
      af3.q[0] = *(const uint4*)(As + o3); af3.q[1] = *(const uint4*)(As + o3 + 16);
    }
    c0 = __builtin_amdgcn_wmma_f32_16x16x32_f16(false, af0.v, false, bf.v, (short)0, c0, false, false);
    c1 = __builtin_amdgcn_wmma_f32_16x16x32_f16(false, af1.v, false, bf.v, (short)0, c1, false, false);
    c2 = __builtin_amdgcn_wmma_f32_16x16x32_f16(false, af2.v, false, bf.v, (short)0, c2, false, false);
    c3 = __builtin_amdgcn_wmma_f32_16x16x32_f16(false, af3.v, false, bf.v, (short)0, c3, false, false);
  }

  if (colOK) {
    float bv = bias[nCol];
    union F8 { v8f v; float f[8]; } r;
    #pragma unroll
    for (int mt = 0; mt < 4; ++mt) {
      r.v = (mt == 0) ? c0 : (mt == 1) ? c1 : (mt == 2) ? c2 : c3;
      #pragma unroll
      for (int rr = 0; rr < 8; ++rr) {
        int row = mBase + mt * 16 + half * 8 + rr;
        out[(size_t)row * V_SIZE + nCol] = r.f[rr] + bv;
      }
    }
  }
}

// --------------------------------------------------------------- launcher ---

extern "C" void kernel_launch(void* const* d_in, const int* in_sizes, int n_in,
                              void* d_out, int out_size, void* d_ws, size_t ws_size,
                              hipStream_t stream) {
  (void)in_sizes; (void)n_in; (void)out_size; (void)ws_size;

  const int*   e1    = (const int*)d_in[0];
  const int*   rel   = (const int*)d_in[1];
  const int*   perm  = (const int*)d_in[2];
  const float* emb   = (const float*)d_in[3];
  const float* fc_w  = (const float*)d_in[4];
  const float* fc_b  = (const float*)d_in[5];
  const float* biasb = (const float*)d_in[6];
  const float* se_w1 = (const float*)d_in[7];
  const float* se_b1 = (const float*)d_in[8];
  const float* se_w2 = (const float*)d_in[9];
  const float* se_b2 = (const float*)d_in[10];
  const float* pag_w = (const float*)d_in[11];
  float* out = (float*)d_out;

  char* w = (char*)d_ws;
  auto bump = [&](size_t bytes) -> char* {
    char* p = w; w += (bytes + 255) & ~(size_t)255; return p;
  };
  float*    xraw   = (float*)bump((size_t)B_DIM * 400 * 4);
  float*    stats1 = (float*)bump(256);
  float*    hbuf   = (float*)bump((size_t)3 * B_DIM * 400 * 4);
  float*    hstats = (float*)bump(256);
  float*    xo     = (float*)bump((size_t)3 * B_DIM * 400 * 4);
  float*    fc_out = (float*)bump((size_t)B_DIM * 200 * 4);
  float*    fstats = (float*)bump(400 * 4);
  _Float16* Amat   = (_Float16*)bump((size_t)B_DIM * KPAD * 2);

  gather_kernel<<<B_DIM, 256, 0, stream>>>(e1, rel, perm, emb, xraw);
  stats1_kernel<<<1, 256, 0, stream>>>(xraw, stats1, B_DIM * 400);
  dwt_kernel<<<dim3(B_DIM, 3), 128, 0, stream>>>(xraw, stats1, hbuf);
  hstats_kernel<<<12, 256, 0, stream>>>(hbuf, hstats);
  se_idwt_kernel<<<dim3(B_DIM, 3), 128, 0, stream>>>(hbuf, hstats, se_w1, se_b1, se_w2, se_b2, xo);
  pag_fc_kernel<<<B_DIM, 256, 0, stream>>>(xo, pag_w, fc_w, fc_b, fc_out);
  fstats_kernel<<<200, 256, 0, stream>>>(fc_out, fstats);
  prep_a_kernel<<<(B_DIM * KPAD + 255) / 256, 256, 0, stream>>>(fc_out, fstats, Amat);

  dim3 ggrid((V_SIZE + 127) / 128, B_DIM / 64);
  gemm_kernel<<<ggrid, 256, 0, stream>>>(Amat, emb, biasb, out);
}
